// TemporalAttnLaye_OptimizedLayer_Kernel_79542794322672
// MI455X (gfx1250) — matrix operand order repaired and checked by reading, not compiled, not run on hardware
//
#include <hip/hip_runtime.h>
#include <hip/hip_bf16.h>
#include <math.h>

#define DN 172
#define DE 172
#define DT 100
#define DOUT 100
#define DH 50
#define ACC_LD 272          // accumulator row stride: [0,100)=attn out, [100,272)=nodeData[dst]
#define NEG_SLOPE 0.2f
#define LN_EPS 1e-5f

typedef __attribute__((ext_vector_type(16))) __bf16 v16bf;
typedef __attribute__((ext_vector_type(8)))  float  v8f;
typedef __attribute__((ext_vector_type(4)))  float  v4f;
typedef __attribute__((ext_vector_type(4)))  __bf16 v4bf;

union FragU { v16bf v; uint4 q[2]; };

// hardware conversion: single cvt -> v_cvt_pk_bf16_f32 (RNE)
__device__ __forceinline__ unsigned short f2bf_bits(float f) {
  union { __bf16 h; unsigned short s; } r; r.h = (__bf16)f; return r.s;
}
// vector conversion: __builtin_convertvector -> two v_cvt_pk_bf16_f32, no repack
__device__ __forceinline__ uint2 pack_bf4(float4 v) {
  v4f f; f.x = v.x; f.y = v.y; f.z = v.z; f.w = v.w;
  union { v4bf b; uint2 u; } r;
  r.b = __builtin_convertvector(f, v4bf);
  return r.u;
}

// order-preserving float atomic max (sign-split trick; mbuf init = -inf)
__device__ __forceinline__ void atomicMaxFloat(float* addr, float v) {
  int iv = __float_as_int(v);
  if (iv >= 0) atomicMax((int*)addr, iv);
  else         atomicMin((unsigned int*)addr, (unsigned int)iv);
}

// ---------------------------------------------------------------------------
// bf16-WMMA GEMM, f32 accumulate.  C[M,N] = act(A@B + bias), B row-major KxN.
// GATHER: A row m is A[idx[m]*lda ...].
// Block 256 thr = 8 waves; tile 128(M) x 32(N); K-step 32.
// Each wave owns a 16x32 strip -> 2 WMMAs per K-step sharing one A fragment.
// Double-buffered LDS (ping-pong): tile t+1 is staged while tile t computes;
// one barrier per K-step. Edge handling via uniform branches + clamped loads.
// ---------------------------------------------------------------------------
#define LDSA_SZ (128 * 40)
#define LDSB_SZ (32 * 40)

template<bool GATHER>
__global__ __launch_bounds__(256)
void gemm_wmma(const float* __restrict__ A, int lda, const int* __restrict__ idx,
               const float* __restrict__ B, const float* __restrict__ bias,
               float* __restrict__ C, int M, int K, int N, int relu)
{
  __shared__ __align__(16) unsigned short ldsA[2 * LDSA_SZ];  // padded stride 40
  __shared__ __align__(16) unsigned short ldsB[2 * LDSB_SZ];  // stored [n][k]

  const int tid  = threadIdx.x;
  const int bm   = blockIdx.x * 128;
  const int bn   = blockIdx.y * 32;
  const int wv   = tid >> 5;
  const int lane = tid & 31;
  const int lrow = lane & 15;
  const int hi   = lane >> 4;

  // ---- hoisted A staging state: thread covers rows arw+32i, k-chunk kc ----
  const int kc  = (tid & 7) << 2;
  const int arw = tid >> 3;
  const float* abase[4];
  float amask[4];
  int aoff[4];
  #pragma unroll
  for (int i = 0; i < 4; ++i) {
    int row = arw + 32 * i;
    int gm  = bm + row;
    bool ok = gm < M;
    int gms = ok ? gm : 0;
    int r   = GATHER ? idx[gms] : gms;
    abase[i] = A + (size_t)r * lda;
    amask[i] = ok ? 1.f : 0.f;
    aoff[i]  = row * 40 + kc;
  }
  const bool m_full = (bm + 128 <= M);  // uniform per block
  // ---- hoisted B staging state ----
  const int bkk = tid >> 3;             // k within tile (0..31)
  const int bnc = (tid & 7) << 2;       // n chunk base
  const bool bn_full = (bn + 32 <= N);  // uniform per block
  const int boff = bnc * 40 + bkk;

  // ---- hoisted fragment LDS offsets ----
  const int paoff  = (wv * 16 + lrow) * 40 + hi * 8;
  const int pb0off = lrow * 40 + hi * 8;
  const int pb1off = (16 + lrow) * 40 + hi * 8;

  v8f acc0 = {0.f,0.f,0.f,0.f,0.f,0.f,0.f,0.f};
  v8f acc1 = {0.f,0.f,0.f,0.f,0.f,0.f,0.f,0.f};

  // stage tile kt into buffer buf
  auto stage = [&](int kt, int buf) {
    unsigned short* la = ldsA + buf * LDSA_SZ;
    unsigned short* lb = ldsB + buf * LDSB_SZ;
    if (kt + 32 <= K) {                         // uniform: full K-tile
      if (m_full) {
        #pragma unroll
        for (int i = 0; i < 4; ++i) {
          float4 v = *(const float4*)(abase[i] + kt + kc);  // aligned b128
          *(uint2*)(la + aoff[i]) = pack_bf4(v);
        }
      } else {
        #pragma unroll
        for (int i = 0; i < 4; ++i) {
          float4 v = *(const float4*)(abase[i] + kt + kc);
          float m  = amask[i];
          v.x *= m; v.y *= m; v.z *= m; v.w *= m;
          *(uint2*)(la + aoff[i]) = pack_bf4(v);
        }
      }
      const float* p = B + (size_t)(kt + bkk) * N;
      if (bn_full) {
        float4 v = *(const float4*)(p + bn + bnc);
        lb[boff]       = f2bf_bits(v.x);
        lb[boff + 40]  = f2bf_bits(v.y);
        lb[boff + 80]  = f2bf_bits(v.z);
        lb[boff + 120] = f2bf_bits(v.w);
      } else {                                  // clamped, branch-free
        #pragma unroll
        for (int t = 0; t < 4; ++t) {
          int n = bn + bnc + t;
          float v = p[n < N ? n : 0];
          lb[boff + t * 40] = f2bf_bits(n < N ? v : 0.f);
        }
      }
    } else {                                    // K-remainder tile (clamped)
      #pragma unroll
      for (int i = 0; i < 4; ++i) {
        const float* rb = abase[i];
        float m = amask[i];
        float4 v;
        float e[4];
        #pragma unroll
        for (int t = 0; t < 4; ++t) {
          int k = kt + kc + t;
          float x = rb[k < K ? k : 0];
          e[t] = (k < K) ? x * m : 0.f;
        }
        v.x = e[0]; v.y = e[1]; v.z = e[2]; v.w = e[3];
        *(uint2*)(la + aoff[i]) = pack_bf4(v);
      }
      int gk = kt + bkk;
      const float* p = B + (size_t)(gk < K ? gk : 0) * N;
      #pragma unroll
      for (int t = 0; t < 4; ++t) {
        int n = bn + bnc + t;
        bool ok = (gk < K) && (n < N);
        float v = p[n < N ? n : 0];
        lb[boff + t * 40] = f2bf_bits(ok ? v : 0.f);
      }
    }
  };

  const int nk = (K + 31) >> 5;
  stage(0, 0);
  __syncthreads();

  for (int t = 0; t < nk; ++t) {
    int buf = t & 1;
    if (t + 1 < nk) stage((t + 1) << 5, buf ^ 1);  // overlap with WMMAs below

    const unsigned short* la = ldsA + buf * LDSA_SZ;
    const unsigned short* lb = ldsB + buf * LDSB_SZ;
    FragU fa, fb0, fb1;
    // A 16x32: lane m=lrow; elems 0-7 -> K=hi*8+0..7, elems 8-15 -> K=16+hi*8+0..7
    fa.q[0] = *(const uint4*)(la + paoff);
    fa.q[1] = *(const uint4*)(la + paoff + 16);
    // B 32x16: lane n=lrow; elems 0-7 -> K=hi*8+0..7, elems 8-15 -> K=16+hi*8+0..7
    fb0.q[0] = *(const uint4*)(lb + pb0off);
    fb0.q[1] = *(const uint4*)(lb + pb0off + 16);
    fb1.q[0] = *(const uint4*)(lb + pb1off);
    fb1.q[1] = *(const uint4*)(lb + pb1off + 16);

    acc0 = __builtin_amdgcn_wmma_f32_16x16x32_bf16(false, fa.v, false, fb0.v,
                                                   (short)0, acc0, false, false);
    acc1 = __builtin_amdgcn_wmma_f32_16x16x32_bf16(false, fa.v, false, fb1.v,
                                                   (short)0, acc1, false, false);
    __syncthreads();   // buf consumed by all waves; buf^1 fully staged
  }

  // D layout: VGPR r -> M = r + 8*hi, N = lrow
  #pragma unroll
  for (int r = 0; r < 8; ++r) {
    int m = bm + wv * 16 + r + hi * 8;
    if (m < M) {
      int n0 = bn + lrow;
      int n1 = bn + 16 + lrow;
      if (n0 < N) {
        float v = acc0[r] + bias[n0];
        if (relu) v = fmaxf(v, 0.f);
        C[(size_t)m * N + n0] = v;
      }
      if (n1 < N) {
        float v = acc1[r] + bias[n1];
        if (relu) v = fmaxf(v, 0.f);
        C[(size_t)m * N + n1] = v;
      }
    }
  }
}

// ---------------------------------------------------------------------------
__global__ void k_time_feat(const float* __restrict__ td, const float* __restrict__ w,
                            const float* __restrict__ b, float* __restrict__ tf, int total)
{
  int i = blockIdx.x * blockDim.x + threadIdx.x;
  if (i >= total) return;
  int t = i / DT, j = i - t * DT;
  tf[i] = cosf(td[t] * w[j] + b[j]);
}

// combo[j] = bqn[j] + bqt[j] + sum_k cos(time_b[k]) * Wqt[k][j]
__global__ void k_combo_bias(const float* __restrict__ time_b, const float* __restrict__ bqn,
                             const float* __restrict__ Wqt, const float* __restrict__ bqt,
                             float* __restrict__ combo)
{
  int j = threadIdx.x;
  if (j >= DOUT) return;
  float s = bqn[j] + bqt[j];
  for (int k = 0; k < DT; ++k) s += cosf(time_b[k]) * Wqt[k * DOUT + j];
  combo[j] = s;
}

// zero attn-accumulator columns [0,100) of the 272-wide buffer; init softmax state
__global__ void k_init(float* __restrict__ accum, float* __restrict__ mbuf,
                       float* __restrict__ sbuf, int num_dst)
{
  int i = blockIdx.x * blockDim.x + threadIdx.x;
  int nAcc = num_dst * DOUT;
  if (i < nAcc) {
    int d = i / DOUT, j = i - d * DOUT;
    accum[(size_t)d * ACC_LD + j] = 0.f;
  }
  if (i < num_dst * 2) { mbuf[i] = -__builtin_inff(); sbuf[i] = 0.f; }
}

// accum[d][100..272) = nodeData[rnids[d]][0..172)
__global__ void k_copy_node(const float* __restrict__ nodeData, const int* __restrict__ rnids,
                            float* __restrict__ accum, int num_dst)
{
  int i = blockIdx.x * blockDim.x + threadIdx.x;
  if (i >= num_dst * DN) return;
  int d = i / DN, j = i - d * DN;
  accum[(size_t)d * ACC_LD + DOUT + j] = nodeData[(size_t)rnids[d] * DN + j];
}

// one wave (32 lanes) per edge: attention logits + segment max
__global__ __launch_bounds__(256)
void k_attn(const float* __restrict__ qdst, const float* __restrict__ kvn,
            const float* __restrict__ kve,  const float* __restrict__ kvt,
            const int* __restrict__ rnids, const int* __restrict__ reids,
            const int* __restrict__ rtids, const int* __restrict__ dstidx,
            float* __restrict__ attn, float* __restrict__ mbuf, int E, int num_dst)
{
  int e = (blockIdx.x << 3) + (threadIdx.x >> 5);
  int lane = threadIdx.x & 31;
  if (e >= E) return;
  int d = dstidx[e];
  size_t ni = (size_t)rnids[num_dst + e] * (2 * DOUT);
  size_t ei = (size_t)reids[e] * (2 * DOUT);
  size_t ti = (size_t)rtids[e] * (2 * DOUT);
  const float* q = qdst + (size_t)d * DOUT;
  float s0 = 0.f, s1 = 0.f;
  for (int j = lane; j < DH; j += 32) {
    float k0 = kvn[ni + j] + kve[ei + j] + kvt[ti + j];
    s0 += q[j] * k0;
    int j1 = DH + j;
    float k1 = kvn[ni + j1] + kve[ei + j1] + kvt[ti + j1];
    s1 += q[j1] * k1;
  }
  #pragma unroll
  for (int off = 16; off > 0; off >>= 1) {
    s0 += __shfl_xor(s0, off, 32);
    s1 += __shfl_xor(s1, off, 32);
  }
  if (lane == 0) {
    float a0 = s0 > 0.f ? s0 : NEG_SLOPE * s0;
    float a1 = s1 > 0.f ? s1 : NEG_SLOPE * s1;
    attn[(size_t)e * 2]     = a0;
    attn[(size_t)e * 2 + 1] = a1;
    atomicMaxFloat(&mbuf[d * 2],     a0);
    atomicMaxFloat(&mbuf[d * 2 + 1], a1);
  }
}

// exp(attn - m[d]) in place + segment sum
__global__ void k_expsum(const int* __restrict__ dstidx, const float* __restrict__ mbuf,
                         float* __restrict__ attn, float* __restrict__ sbuf, int E)
{
  int t = blockIdx.x * blockDim.x + threadIdx.x;
  if (t >= 2 * E) return;
  int e = t >> 1, h = t & 1;
  int d = dstidx[e];
  float ev = expf(attn[t] - mbuf[d * 2 + h]);
  attn[t] = ev;
  atomicAdd(&sbuf[d * 2 + h], ev);
}

// one wave per edge: accum[d][0..100) += a_h * V
__global__ __launch_bounds__(256)
void k_scatter(const float* __restrict__ kvn, const float* __restrict__ kve,
               const float* __restrict__ kvt,
               const int* __restrict__ rnids, const int* __restrict__ reids,
               const int* __restrict__ rtids, const int* __restrict__ dstidx,
               const float* __restrict__ attn, const float* __restrict__ sbuf,
               float* __restrict__ accum, int E, int num_dst)
{
  int e = (blockIdx.x << 3) + (threadIdx.x >> 5);
  int lane = threadIdx.x & 31;
  if (e >= E) return;
  int d = dstidx[e];
  size_t ni = (size_t)rnids[num_dst + e] * (2 * DOUT) + DOUT;
  size_t ei = (size_t)reids[e] * (2 * DOUT) + DOUT;
  size_t ti = (size_t)rtids[e] * (2 * DOUT) + DOUT;
  float a0 = attn[(size_t)e * 2]     / sbuf[d * 2];
  float a1 = attn[(size_t)e * 2 + 1] / sbuf[d * 2 + 1];
  for (int j = lane; j < DOUT; j += 32) {
    float v = kvn[ni + j] + kve[ei + j] + kvt[ti + j];
    float w = (j < DH) ? a0 : a1;
    atomicAdd(&accum[(size_t)d * ACC_LD + j], v * w);
  }
}

// one wave per row, in-place LayerNorm over DOUT=100
__global__ __launch_bounds__(256)
void k_layernorm(float* __restrict__ out, const float* __restrict__ g,
                 const float* __restrict__ b, int M)
{
  int w = (blockIdx.x * 256 + threadIdx.x) >> 5;
  int lane = threadIdx.x & 31;
  if (w >= M) return;
  float* row = out + (size_t)w * DOUT;
  float x[4];
  float s = 0.f, ss = 0.f;
  #pragma unroll
  for (int i = 0; i < 4; ++i) {
    int j = lane + i * 32;
    float v = (j < DOUT) ? row[j] : 0.f;
    x[i] = v; s += v; ss += v * v;
  }
  #pragma unroll
  for (int off = 16; off > 0; off >>= 1) {
    s  += __shfl_xor(s,  off, 32);
    ss += __shfl_xor(ss, off, 32);
  }
  float mean = s * (1.f / DOUT);
  float var  = ss * (1.f / DOUT) - mean * mean;
  float rstd = rsqrtf(var + LN_EPS);
  #pragma unroll
  for (int i = 0; i < 4; ++i) {
    int j = lane + i * 32;
    if (j < DOUT) row[j] = (x[i] - mean) * rstd * g[j] + b[j];
  }
}

// ---------------------------------------------------------------------------
extern "C" void kernel_launch(void* const* d_in, const int* in_sizes, int n_in,
                              void* d_out, int out_size, void* d_ws, size_t ws_size,
                              hipStream_t stream)
{
  const float* nodeData = (const float*)d_in[0];
  const float* efeat    = (const float*)d_in[1];
  const float* utd      = (const float*)d_in[2];
  const int*   rnids    = (const int*)d_in[3];
  const int*   reids    = (const int*)d_in[4];
  const int*   rtids    = (const int*)d_in[5];
  const int*   dstidx   = (const int*)d_in[6];
  /* d_in[7] = num_dst scalar (derived from sizes instead) */
  const float* time_w = (const float*)d_in[8];
  const float* time_b = (const float*)d_in[9];
  const float* Wqn  = (const float*)d_in[10];
  const float* bqn  = (const float*)d_in[11];
  const float* Wqt  = (const float*)d_in[12];
  const float* bqt  = (const float*)d_in[13];
  const float* Wkvn = (const float*)d_in[14];
  const float* bkvn = (const float*)d_in[15];
  const float* Wkve = (const float*)d_in[16];
  const float* bkve = (const float*)d_in[17];
  const float* Wkvt = (const float*)d_in[18];
  const float* bkvt = (const float*)d_in[19];
  const float* Wout = (const float*)d_in[20];
  const float* bout = (const float*)d_in[21];
  const float* ln_g = (const float*)d_in[22];
  const float* ln_b = (const float*)d_in[23];

  const int E       = in_sizes[4];
  const int num_dst = in_sizes[3] - E;
  const int U_T     = in_sizes[2];
  const int U_N     = in_sizes[0] / DN;
  const int U_E     = in_sizes[1] / DE;

  float* ws = (float*)d_ws;
  size_t off = 0;
  float* tf    = ws + off; off += (size_t)U_T * DT;
  float* qdst  = ws + off; off += (size_t)num_dst * DOUT;
  float* kvn   = ws + off; off += (size_t)U_N * 2 * DOUT;
  float* kve   = ws + off; off += (size_t)U_E * 2 * DOUT;
  float* kvt   = ws + off; off += (size_t)U_T * 2 * DOUT;
  float* combo = ws + off; off += 128;
  float* mbuf  = ws + off; off += (size_t)num_dst * 2;
  float* sbuf  = ws + off; off += (size_t)num_dst * 2;
  float* attn  = ws + off; off += (size_t)E * 2;
  float* accum = ws + off; off += (size_t)num_dst * ACC_LD;

  float* out = (float*)d_out;
  dim3 blk(256);
  auto gemm_grid = [](int M, int N) {
    return dim3((unsigned)((M + 127) / 128), (unsigned)((N + 31) / 32));
  };

  // 1) time features cos(td*w+b)
  { int total = U_T * DT;
    k_time_feat<<<(total + 255) / 256, 256, 0, stream>>>(utd, time_w, time_b, tf, total); }
  // 2) fused q bias: bqn + bqt + cos(time_b)@Wqt
  k_combo_bias<<<1, 128, 0, stream>>>(time_b, bqn, Wqt, bqt, combo);
  // 3) init softmax state + accumulator cols [0,100); fill cols [100,272) with gathered nodeData
  { int nAcc = num_dst * DOUT;
    k_init<<<(nAcc + 255) / 256, 256, 0, stream>>>(accum, mbuf, sbuf, num_dst); }
  { int total = num_dst * DN;
    k_copy_node<<<(total + 255) / 256, 256, 0, stream>>>(nodeData, rnids, accum, num_dst); }
  // 4) WMMA GEMMs (unique-entity projections)
  gemm_wmma<true><<<gemm_grid(num_dst, DOUT), blk, 0, stream>>>(
      nodeData, DN, rnids, Wqn, combo, qdst, num_dst, DN, DOUT, 0);
  gemm_wmma<false><<<gemm_grid(U_N, 2 * DOUT), blk, 0, stream>>>(
      nodeData, DN, nullptr, Wkvn, bkvn, kvn, U_N, DN, 2 * DOUT, 0);
  gemm_wmma<false><<<gemm_grid(U_E, 2 * DOUT), blk, 0, stream>>>(
      efeat, DE, nullptr, Wkve, bkve, kve, U_E, DE, 2 * DOUT, 0);
  gemm_wmma<false><<<gemm_grid(U_T, 2 * DOUT), blk, 0, stream>>>(
      tf, DT, nullptr, Wkvt, bkvt, kvt, U_T, DT, 2 * DOUT, 0);
  // 5) edge logits + segment max
  k_attn<<<(E + 7) / 8, 256, 0, stream>>>(qdst, kvn, kve, kvt, rnids, reids, rtids,
                                          dstidx, attn, mbuf, E, num_dst);
  // 6) exp + segment sum
  { int total = 2 * E;
    k_expsum<<<(total + 255) / 256, 256, 0, stream>>>(dstidx, mbuf, attn, sbuf, E); }
  // 7) weighted V scatter-add
  k_scatter<<<(E + 7) / 8, 256, 0, stream>>>(kvn, kve, kvt, rnids, reids, rtids,
                                             dstidx, attn, sbuf, accum, E, num_dst);
  // 8) output GEMM: accum(272-wide, already concat) @ Wout + bout, ReLU
  gemm_wmma<false><<<gemm_grid(num_dst, DOUT), blk, 0, stream>>>(
      accum, ACC_LD, nullptr, Wout, bout, out, num_dst, DOUT + DN, DOUT, 1);
  // 9) LayerNorm in place
  k_layernorm<<<((num_dst * 32) + 255) / 256, 256, 0, stream>>>(out, ln_g, ln_b, num_dst);
}
